// RelationalGraphNeuralNetwork_21973052686564
// MI455X (gfx1250) — matrix operand order repaired and checked by reading, not compiled, not run on hardware
//
#include <hip/hip_runtime.h>
#include <hip/hip_bf16.h>

typedef __attribute__((ext_vector_type(16))) _Float16 v16h;
typedef __attribute__((ext_vector_type(4)))  _Float16 v4h;
typedef __attribute__((ext_vector_type(8)))  float    v8f;

#define DD 64
#define NGRAPHS 100

// mish(x) = x*tanh(softplus(x)) = x*(p^2-1)/(p^2+1), p = 1+e^x.
// One v_exp_f32 + one v_rcp_f32; correct tails (p^2->inf => mish->x).
__device__ __forceinline__ float mish_f(float x) {
  float p = 1.0f + __expf(x);
  float q = p * p;
  float t = __builtin_amdgcn_rcpf(q + 1.0f);
  return x * (q - 1.0f) * t;
}

// Compute one 16x16 f32 C tile: A (LDS f16, 16 rows, leading dim lda),
// B = f16 weights [K][ldw] row-major in global, column block nb, reduce over K.
__device__ __forceinline__ v8f wmma_tile(const _Float16* A, int lda,
                                         const _Float16* __restrict__ W, int ldw,
                                         int nb, int K) {
  const int lane = threadIdx.x & 31;
  const int m = lane & 15;
  const bool hi = lane >= 16;
  v8f c = {0.f, 0.f, 0.f, 0.f, 0.f, 0.f, 0.f, 0.f};
  for (int kb = 0; kb < K; kb += 32) {
    v16h a, b;
    // A fragment: lane<16 holds K kb+{0..7,16..23}; lane>=16 holds kb+{8..15,24..31}
    const _Float16* ap = A + m * lda + kb + (hi ? 8 : 0);
#pragma unroll
    for (int i = 0; i < 8; ++i) { a[i] = ap[i]; a[i + 8] = ap[i + 16]; }
    // B fragment: column n = nb+m; lane<16 holds K kb+{0..7,16..23}; lane>=16 holds kb+{8..15,24..31}
    const _Float16* bp = W + (long)(kb + (hi ? 8 : 0)) * ldw + nb + m;
#pragma unroll
    for (int i = 0; i < 8; ++i) { b[i] = bp[(long)i * ldw]; b[i + 8] = bp[(long)(i + 16) * ldw]; }
    c = __builtin_amdgcn_wmma_f32_16x16x32_f16(false, a, false, b, (short)0, c,
                                               false, false);
  }
  return c;
}

// ---------------- utility kernels ----------------
__global__ void zero_f32_kernel(float* __restrict__ p, long n) {
  long i = (long)blockIdx.x * blockDim.x + threadIdx.x;
  if (i < n) p[i] = 0.f;
}

__global__ void cvt_f16_kernel(const float* __restrict__ s, _Float16* __restrict__ d, int n) {
  int i = blockIdx.x * blockDim.x + threadIdx.x;
  if (i < n) d[i] = (_Float16)s[i];
}

__global__ void count_kernel(const int* __restrict__ idx, int M, float* __restrict__ c) {
  int i = blockIdx.x * blockDim.x + threadIdx.x;
  if (i < M) atomicAdd(&c[idx[i]], 1.0f);
}

__global__ void inv_count_kernel(float* __restrict__ c, int N) {
  int i = blockIdx.x * blockDim.x + threadIdx.x;
  if (i < N) c[i] = 1.0f / fmaxf(c[i], 1.0f);
}

// Per-graph sums with run-length accumulation over 16-node chunks:
// ~16x fewer atomics than per-element, still driven by graph_ids.
__global__ void gsum_kernel(const float* __restrict__ x, const int* __restrict__ gids,
                            float* __restrict__ gsum, int N) {
  const int CH = 16;
  int nchunks = (N + CH - 1) / CH;
  long t = (long)blockIdx.x * blockDim.x + threadIdx.x;
  if (t >= (long)nchunks * DD) return;
  int d = (int)(t % DD);
  int chunk = (int)(t / DD);
  int n0 = chunk * CH, n1 = min(n0 + CH, N);
  float acc = 0.f;
  int cur = gids[n0];
  for (int n = n0; n < n1; ++n) {
    int g = gids[n];
    if (g != cur) { atomicAdd(&gsum[cur * DD + d], acc); acc = 0.f; cur = g; }
    acc += x[(long)n * DD + d];
  }
  atomicAdd(&gsum[cur * DD + d], acc);
}

// ---------------- relation message kernel ----------------
template <int ARITY>
__global__ void msg_kernel(const float* __restrict__ x, const int* __restrict__ idx, int E,
                           const _Float16* __restrict__ W1, const float* __restrict__ b1,
                           const _Float16* __restrict__ W2, const float* __restrict__ b2,
                           float* __restrict__ agg) {
  constexpr int K = ARITY * DD;
  constexpr int LDW = K + 8;
  __shared__ _Float16 s_in[4][16][LDW];
  __shared__ _Float16 s_hid[4][16][LDW];
  const int wid = threadIdx.x >> 5, lane = threadIdx.x & 31;
  const int tile = blockIdx.x * 4 + wid;
  const int e0 = tile * 16;
  const int m = lane & 15;
  const bool hi = lane >= 16;

  // vectorized gather: float4 from node rows -> 4-half packets in LDS
  for (int i = lane; i < 16 * (K / 4); i += 32) {
    int row = i / (K / 4), c4 = (i % (K / 4)) * 4;
    int e = min(e0 + row, E - 1);
    int a = c4 >> 6, d = c4 & 63;
    float4 v = *(const float4*)&x[(long)idx[e * ARITY + a] * DD + d];
    v4h h; h[0] = (_Float16)v.x; h[1] = (_Float16)v.y; h[2] = (_Float16)v.z; h[3] = (_Float16)v.w;
    *(v4h*)&s_in[wid][row][c4] = h;
  }
  __syncthreads();

  for (int nt = 0; nt < K / 16; ++nt) {
    int nb = nt * 16;
    v8f c = wmma_tile(&s_in[wid][0][0], LDW, W1, K, nb, K);
    float bias = b1[nb + m];
#pragma unroll
    for (int v = 0; v < 8; ++v)
      s_hid[wid][v + (hi ? 8 : 0)][nb + m] = (_Float16)mish_f(c[v] + bias);
  }
  __syncthreads();

  for (int nt = 0; nt < K / 16; ++nt) {
    int nb = nt * 16;
    v8f c = wmma_tile(&s_hid[wid][0][0], LDW, W2, K, nb, K);
    float bias = b2[nb + m];
    int a = (nb + m) >> 6, d = (nb + m) & 63;
#pragma unroll
    for (int v = 0; v < 8; ++v) {
      int e = e0 + v + (hi ? 8 : 0);
      if (e < E) atomicAdd(&agg[(long)idx[e * ARITY + a] * DD + d], c[v] + bias);
    }
  }
}

// ---------------- dense 64->64->64 MLP over R rows (global readout) ----------------
__global__ void rows_mlp64_kernel(const float* __restrict__ in, int R,
                                  const _Float16* __restrict__ W1, const float* __restrict__ b1,
                                  const _Float16* __restrict__ W2, const float* __restrict__ b2,
                                  float* __restrict__ out) {
  constexpr int K = 64, LDW = K + 8;
  __shared__ _Float16 s_in[4][16][LDW];
  __shared__ _Float16 s_hid[4][16][LDW];
  const int wid = threadIdx.x >> 5, lane = threadIdx.x & 31;
  const int r0 = (blockIdx.x * 4 + wid) * 16;
  const int m = lane & 15;
  const bool hi = lane >= 16;

  for (int i = lane; i < 16 * (K / 4); i += 32) {
    int row = i / (K / 4), c4 = (i % (K / 4)) * 4;
    int r = min(r0 + row, R - 1);
    float4 v = *(const float4*)&in[(long)r * K + c4];
    v4h h; h[0] = (_Float16)v.x; h[1] = (_Float16)v.y; h[2] = (_Float16)v.z; h[3] = (_Float16)v.w;
    *(v4h*)&s_in[wid][row][c4] = h;
  }
  __syncthreads();
  for (int nt = 0; nt < K / 16; ++nt) {
    int nb = nt * 16;
    v8f c = wmma_tile(&s_in[wid][0][0], LDW, W1, K, nb, K);
    float bias = b1[nb + m];
#pragma unroll
    for (int v = 0; v < 8; ++v)
      s_hid[wid][v + (hi ? 8 : 0)][nb + m] = (_Float16)mish_f(c[v] + bias);
  }
  __syncthreads();
  for (int nt = 0; nt < K / 16; ++nt) {
    int nb = nt * 16;
    v8f c = wmma_tile(&s_hid[wid][0][0], LDW, W2, K, nb, K);
    float bias = b2[nb + m];
#pragma unroll
    for (int v = 0; v < 8; ++v) {
      int r = r0 + v + (hi ? 8 : 0);
      if (r < R) out[(long)r * K + nb + m] = c[v] + bias;
    }
  }
}

// ---------------- fused node update ----------------
// nxt = LN(u_mlp([x|agg/cnt])), gmsg = LN(gu_mlp([x|glob[g]])), xout = x + nxt + gmsg
__global__ void node_update_kernel(const float* __restrict__ x, const float* __restrict__ agg,
                                   const float* __restrict__ invc, const float* __restrict__ glob,
                                   const int* __restrict__ gids,
                                   const _Float16* __restrict__ uW1, const float* __restrict__ ub1,
                                   const _Float16* __restrict__ uW2, const float* __restrict__ ub2,
                                   const _Float16* __restrict__ gW1, const float* __restrict__ gb1,
                                   const _Float16* __restrict__ gW2, const float* __restrict__ gb2,
                                   const float* __restrict__ lng, const float* __restrict__ lnb,
                                   float* __restrict__ xout, int N) {
  constexpr int K = 128, LDW = K + 8;
  __shared__ _Float16 s_in[2][16][LDW];
  __shared__ _Float16 s_hid[2][16][LDW];
  __shared__ float s_nxt[2][16][DD];
  __shared__ float s_gm[2][16][DD];
  __shared__ float s_mu1[2][16], s_r1[2][16], s_mu2[2][16], s_r2[2][16];
  const int wid = threadIdx.x >> 5, lane = threadIdx.x & 31;
  const int n0 = (blockIdx.x * 2 + wid) * 16;
  const int m = lane & 15;
  const bool hi = lane >= 16;

  for (int pass = 0; pass < 2; ++pass) {
    for (int i = lane; i < 16 * (K / 4); i += 32) {
      int row = i / (K / 4), c4 = (i % (K / 4)) * 4;
      int n = min(n0 + row, N - 1);
      float4 v;
      if (c4 < DD) {
        v = *(const float4*)&x[(long)n * DD + c4];
      } else {
        int d = c4 - DD;
        if (pass == 0) {
          v = *(const float4*)&agg[(long)n * DD + d];
          float ic = invc[n];
          v.x *= ic; v.y *= ic; v.z *= ic; v.w *= ic;
        } else {
          v = *(const float4*)&glob[(long)gids[n] * DD + d];
        }
      }
      v4h h; h[0] = (_Float16)v.x; h[1] = (_Float16)v.y; h[2] = (_Float16)v.z; h[3] = (_Float16)v.w;
      *(v4h*)&s_in[wid][row][c4] = h;
    }
    __syncthreads();
    const _Float16* W1 = pass ? gW1 : uW1; const float* B1 = pass ? gb1 : ub1;
    const _Float16* W2 = pass ? gW2 : uW2; const float* B2 = pass ? gb2 : ub2;
    for (int nt = 0; nt < K / 16; ++nt) {
      int nb = nt * 16;
      v8f c = wmma_tile(&s_in[wid][0][0], LDW, W1, K, nb, K);
      float bias = B1[nb + m];
#pragma unroll
      for (int v = 0; v < 8; ++v)
        s_hid[wid][v + (hi ? 8 : 0)][nb + m] = (_Float16)mish_f(c[v] + bias);
    }
    __syncthreads();
    float* outT = pass ? &s_gm[wid][0][0] : &s_nxt[wid][0][0];
    for (int nt = 0; nt < DD / 16; ++nt) {
      int nb = nt * 16;
      v8f c = wmma_tile(&s_hid[wid][0][0], LDW, W2, DD, nb, K);
      float bias = B2[nb + m];
#pragma unroll
      for (int v = 0; v < 8; ++v)
        outT[(v + (hi ? 8 : 0)) * DD + nb + m] = c[v] + bias;
    }
    __syncthreads();
  }

  // stats by 16 lanes
  if (lane < 16) {
    float mu1 = 0.f, mu2 = 0.f, s1 = 0.f, s2 = 0.f;
    for (int d = 0; d < DD; ++d) {
      float a = s_nxt[wid][lane][d], b = s_gm[wid][lane][d];
      mu1 += a; mu2 += b; s1 += a * a; s2 += b * b;
    }
    mu1 *= (1.f / DD); mu2 *= (1.f / DD);
    s_mu1[wid][lane] = mu1; s_mu2[wid][lane] = mu2;
    s_r1[wid][lane] = rsqrtf(s1 * (1.f / DD) - mu1 * mu1 + 1e-5f);
    s_r2[wid][lane] = rsqrtf(s2 * (1.f / DD) - mu2 * mu2 + 1e-5f);
  }
  __syncthreads();

  // coalesced LN + residual write by all 32 lanes
  for (int i = lane; i < 16 * DD; i += 32) {
    int row = i >> 6, d = i & 63;
    int n = n0 + row;
    if (n < N) {
      float ln1 = (s_nxt[wid][row][d] - s_mu1[wid][row]) * s_r1[wid][row] * lng[d] + lnb[d];
      float ln2 = (s_gm[wid][row][d] - s_mu2[wid][row]) * s_r2[wid][row] * lng[d] + lnb[d];
      xout[(long)n * DD + d] = x[(long)n * DD + d] + ln1 + ln2;
    }
  }
}

// ---------------- host orchestration ----------------
static inline int cdiv(long a, long b) { return (int)((a + b - 1) / b); }

extern "C" void kernel_launch(void* const* d_in, const int* in_sizes, int n_in,
                              void* d_out, int out_size, void* d_ws, size_t ws_size,
                              hipStream_t stream) {
  (void)n_in; (void)out_size; (void)ws_size;
  const float* m1W1 = (const float*)d_in[0];  const float* m1b1 = (const float*)d_in[1];
  const float* m1W2 = (const float*)d_in[2];  const float* m1b2 = (const float*)d_in[3];
  const float* m2W1 = (const float*)d_in[4];  const float* m2b1 = (const float*)d_in[5];
  const float* m2W2 = (const float*)d_in[6];  const float* m2b2 = (const float*)d_in[7];
  const float* m3W1 = (const float*)d_in[8];  const float* m3b1 = (const float*)d_in[9];
  const float* m3W2 = (const float*)d_in[10]; const float* m3b2 = (const float*)d_in[11];
  const float* uW1  = (const float*)d_in[12]; const float* ub1  = (const float*)d_in[13];
  const float* uW2  = (const float*)d_in[14]; const float* ub2  = (const float*)d_in[15];
  const float* lng  = (const float*)d_in[16]; const float* lnb  = (const float*)d_in[17];
  const float* grW1 = (const float*)d_in[18]; const float* grb1 = (const float*)d_in[19];
  const float* grW2 = (const float*)d_in[20]; const float* grb2 = (const float*)d_in[21];
  const float* guW1 = (const float*)d_in[22]; const float* gub1 = (const float*)d_in[23];
  const float* guW2 = (const float*)d_in[24]; const float* gub2 = (const float*)d_in[25];
  const int* idx1 = (const int*)d_in[26];
  const int* idx2 = (const int*)d_in[27];
  const int* idx3 = (const int*)d_in[28];
  const int* gids = (const int*)d_in[29];
  const int E1 = in_sizes[26] / 2, E2 = in_sizes[27] / 2, E3 = in_sizes[28];
  const int N = in_sizes[29];

  // workspace carve (256B aligned)
  char* p = (char*)d_ws;
  auto carve = [&](size_t bytes) { void* r = (void*)p; p += (bytes + 255) & ~(size_t)255; return r; };
  float* xA   = (float*)carve((size_t)N * DD * 4);
  float* xB   = (float*)carve((size_t)N * DD * 4);
  float* agg  = (float*)carve((size_t)N * DD * 4);
  float* invc = (float*)carve((size_t)N * 4);
  float* gsum = (float*)carve((size_t)NGRAPHS * DD * 4);
  float* glob = (float*)carve((size_t)NGRAPHS * DD * 4);
  struct WCvt { const float* src; int n; } wl[12] = {
      {m1W1, 128 * 128}, {m1W2, 128 * 128}, {m2W1, 128 * 128}, {m2W2, 128 * 128},
      {m3W1, 64 * 64},   {m3W2, 64 * 64},   {uW1, 128 * 128},  {uW2, 128 * 64},
      {grW1, 64 * 64},   {grW2, 64 * 64},   {guW1, 128 * 128}, {guW2, 128 * 64}};
  _Float16* wh[12];
  for (int i = 0; i < 12; ++i) {
    wh[i] = (_Float16*)carve((size_t)wl[i].n * 2);
    cvt_f16_kernel<<<cdiv(wl[i].n, 256), 256, 0, stream>>>(wl[i].src, wh[i], wl[i].n);
  }
  _Float16 *m1W1h = wh[0], *m1W2h = wh[1], *m2W1h = wh[2], *m2W2h = wh[3];
  _Float16 *m3W1h = wh[4], *m3W2h = wh[5], *uW1h = wh[6], *uW2h = wh[7];
  _Float16 *grW1h = wh[8], *grW2h = wh[9], *guW1h = wh[10], *guW2h = wh[11];

  zero_f32_kernel<<<cdiv(N, 256), 256, 0, stream>>>(invc, N);
  count_kernel<<<cdiv(in_sizes[26], 256), 256, 0, stream>>>(idx1, in_sizes[26], invc);
  count_kernel<<<cdiv(in_sizes[27], 256), 256, 0, stream>>>(idx2, in_sizes[27], invc);
  count_kernel<<<cdiv(in_sizes[28], 256), 256, 0, stream>>>(idx3, in_sizes[28], invc);
  inv_count_kernel<<<cdiv(N, 256), 256, 0, stream>>>(invc, N);

  zero_f32_kernel<<<cdiv((long)N * DD, 256), 256, 0, stream>>>(xA, (long)N * DD);

  float* bufs[2] = {xA, xB};
  for (int layer = 0; layer < 6; ++layer) {
    float* xc = bufs[layer & 1];
    float* xn = (layer == 5) ? (float*)d_out : bufs[(layer + 1) & 1];

    zero_f32_kernel<<<cdiv((long)N * DD, 256), 256, 0, stream>>>(agg, (long)N * DD);
    msg_kernel<2><<<cdiv(cdiv(E1, 16), 4), 128, 0, stream>>>(xc, idx1, E1, m1W1h, m1b1, m1W2h, m1b2, agg);
    msg_kernel<2><<<cdiv(cdiv(E2, 16), 4), 128, 0, stream>>>(xc, idx2, E2, m2W1h, m2b1, m2W2h, m2b2, agg);
    msg_kernel<1><<<cdiv(cdiv(E3, 16), 4), 128, 0, stream>>>(xc, idx3, E3, m3W1h, m3b1, m3W2h, m3b2, agg);

    zero_f32_kernel<<<cdiv((long)NGRAPHS * DD, 256), 256, 0, stream>>>(gsum, (long)NGRAPHS * DD);
    gsum_kernel<<<cdiv((long)cdiv(N, 16) * DD, 256), 256, 0, stream>>>(xc, gids, gsum, N);
    rows_mlp64_kernel<<<cdiv(cdiv(NGRAPHS, 16), 4), 128, 0, stream>>>(gsum, NGRAPHS, grW1h, grb1, grW2h, grb2, glob);

    node_update_kernel<<<cdiv(cdiv(N, 16), 2), 64, 0, stream>>>(
        xc, agg, invc, glob, gids, uW1h, ub1, uW2h, ub2, guW1h, gub1, guW2h, gub2,
        lng, lnb, xn, N);
  }
}
